// VQVAE_58798102282861
// MI455X (gfx1250) — matrix operand order repaired and checked by reading, not compiled, not run on hardware
//
#include <hip/hip_runtime.h>

typedef __attribute__((ext_vector_type(16))) __bf16 v16bf;
typedef __attribute__((ext_vector_type(8)))  float  v8f;

// ---------------------------------------------------------------------------
// Weight transform: fp32 HWIO [K][Cout]  ->  bf16 [Cout][K]  (transpose+cvt)
// Also used for the VQ codebook: emb [D][Kcb] -> embT [Kcb][D].
// ---------------------------------------------------------------------------
__global__ void transform_w(const float* __restrict__ w, __bf16* __restrict__ wt,
                            int K, int Cout)
{
    long total = (long)K * Cout;
    long i = blockIdx.x * (long)blockDim.x + threadIdx.x;
    if (i >= total) return;
    int  n = (int)(i % Cout);
    long k = i / Cout;
    wt[(long)n * K + k] = (__bf16)w[i];
}

// ---------------------------------------------------------------------------
// Implicit-GEMM conv, Cin % 32 == 0 fast path.
// One wave computes a 16(M) x 64(N) register tile: the A fragment (4x float4
// loads + bf16 cvt) is reused by FOUR v_wmma_f32_16x16x32_bf16 per K-chunk.
// B fragments are contiguous 32-byte v16bf loads from pre-transposed weights.
// ---------------------------------------------------------------------------
__global__ void conv_wmma_al(const float* __restrict__ in, const __bf16* __restrict__ wt,
                             const float* __restrict__ bias, const float* __restrict__ res,
                             float* __restrict__ out,
                             int B, int H, int W, int Cin, int Ho, int Wo, int Cout,
                             int KH, int KW, int stride, int pad)
{
    const int waveId = threadIdx.x >> 5;
    const int lane   = threadIdx.x & 31;
    const int M = B * Ho * Wo;
    const int mtiles = (M + 15) >> 4;
    const int ntiles = (Cout + 63) >> 6;          // 64-wide N macro-tiles
    const int tile = blockIdx.x * (blockDim.x >> 5) + waveId;
    if (tile >= mtiles * ntiles) return;          // wave-uniform: EXEC stays all-1s

    const int nt = tile % ntiles;
    const int mt = tile / ntiles;
    const int mBase = mt << 4;
    const int nBase = nt << 6;
    const int hi = lane >> 4;
    const int lr = lane & 15;

    const int m  = mBase + lr;
    const bool mvalid = (m < M);
    const int mm = mvalid ? m : (M - 1);
    const int ox = mm % Wo;
    const int t1 = mm / Wo;
    const int oy = t1 % Ho;
    const int b  = t1 / Ho;

    const int Ktot = KH * KW * Cin;
    const __bf16* wrow[4];
    bool nv[4];
#pragma unroll
    for (int j = 0; j < 4; ++j) {
        int n = nBase + j * 16 + lr;
        nv[j] = (n < Cout);
        wrow[j] = wt + (nv[j] ? (long)n * Ktot : 0);
    }

    v8f acc[4] = {};

    for (int ky = 0; ky < KH; ++ky) {
        const int iy = oy * stride + ky - pad;
        for (int kx = 0; kx < KW; ++kx) {
            const int ix = ox * stride + kx - pad;
            const bool ok = mvalid && iy >= 0 && iy < H && ix >= 0 && ix < W;
            const float* src = in + (((long)b * H + iy) * W + ix) * Cin;
            const int kpatch = (ky * KW + kx) * Cin;
            // prefetch the next kernel-window column's activations
            if (ok && kx + 1 < KW)
                __builtin_prefetch(src + Cin, 0, 1);
            for (int ci0 = 0; ci0 < Cin; ci0 += 32) {
                float4 p0 = make_float4(0.f, 0.f, 0.f, 0.f), p1 = p0, p2 = p0, p3 = p0;
                if (ok) {
                    p0 = *(const float4*)(src + ci0 + hi * 16);
                    p1 = *(const float4*)(src + ci0 + hi * 16 + 4);
                    p2 = *(const float4*)(src + ci0 + hi * 16 + 8);
                    p3 = *(const float4*)(src + ci0 + hi * 16 + 12);
                }
                v16bf a;
                a[0]=(__bf16)p0.x;  a[1]=(__bf16)p0.y;  a[2]=(__bf16)p0.z;  a[3]=(__bf16)p0.w;
                a[4]=(__bf16)p1.x;  a[5]=(__bf16)p1.y;  a[6]=(__bf16)p1.z;  a[7]=(__bf16)p1.w;
                a[8]=(__bf16)p2.x;  a[9]=(__bf16)p2.y;  a[10]=(__bf16)p2.z; a[11]=(__bf16)p2.w;
                a[12]=(__bf16)p3.x; a[13]=(__bf16)p3.y; a[14]=(__bf16)p3.z; a[15]=(__bf16)p3.w;

                const int koff = kpatch + ci0 + hi * 16;
                v16bf b0 = {}, b1 = {}, b2 = {}, b3 = {};
                if (nv[0]) b0 = *(const v16bf*)(wrow[0] + koff);
                if (nv[1]) b1 = *(const v16bf*)(wrow[1] + koff);
                if (nv[2]) b2 = *(const v16bf*)(wrow[2] + koff);
                if (nv[3]) b3 = *(const v16bf*)(wrow[3] + koff);

                acc[0] = __builtin_amdgcn_wmma_f32_16x16x32_bf16(false, a, false, b0, (short)0, acc[0], false, false);
                acc[1] = __builtin_amdgcn_wmma_f32_16x16x32_bf16(false, a, false, b1, (short)0, acc[1], false, false);
                acc[2] = __builtin_amdgcn_wmma_f32_16x16x32_bf16(false, a, false, b2, (short)0, acc[2], false, false);
                acc[3] = __builtin_amdgcn_wmma_f32_16x16x32_bf16(false, a, false, b3, (short)0, acc[3], false, false);
            }
        }
    }

#pragma unroll
    for (int j = 0; j < 4; ++j) {
        int no = nBase + j * 16 + lr;
        if (no < Cout) {
            float bs = bias[no];
#pragma unroll
            for (int r = 0; r < 8; ++r) {
                int mo = mBase + r + hi * 8;
                if (mo < M) {
                    float v = acc[j][r] + bs;
                    if (res) v += res[(long)mo * Cout + no];
                    out[(long)mo * Cout + no] = v;
                }
            }
        }
    }
}

// ---------------------------------------------------------------------------
// Generic (scalar-load) conv: only used for the Cin=3 input conv (Ktot=27).
// ---------------------------------------------------------------------------
__global__ void conv_wmma_gen(const float* __restrict__ in, const float* __restrict__ w,
                              const float* __restrict__ bias, float* __restrict__ out,
                              int B, int H, int W, int Cin, int Ho, int Wo, int Cout,
                              int KH, int KW, int stride, int pad)
{
    const int waveId = threadIdx.x >> 5;
    const int lane   = threadIdx.x & 31;
    const int M = B * Ho * Wo;
    const int mtiles = (M + 15) >> 4;
    const int ntiles = (Cout + 15) >> 4;
    const int tile = blockIdx.x * (blockDim.x >> 5) + waveId;
    if (tile >= mtiles * ntiles) return;

    const int nt = tile % ntiles;
    const int mt = tile / ntiles;
    const int mBase = mt << 4;
    const int nBase = nt << 4;
    const int hi = lane >> 4;
    const int lr = lane & 15;

    const int m  = mBase + lr;
    const bool mvalid = (m < M);
    const int mm = mvalid ? m : (M - 1);
    const int ox = mm % Wo;
    const int t1 = mm / Wo;
    const int oy = t1 % Ho;
    const int b  = t1 / Ho;
    const int n  = nBase + lr;

    const int Ktot = KH * KW * Cin;
    v8f acc = {};

    for (int k0 = 0; k0 < Ktot; k0 += 32) {
        v16bf a, bm;
#pragma unroll
        for (int e = 0; e < 16; ++e) {
            int ka = k0 + ((e < 8) ? e : e + 8) + hi * 8;
            float av = 0.0f;
            if (mvalid && ka < Ktot) {
                int ci = ka % Cin;
                int p  = ka / Cin;
                int kx = p % KW;
                int ky = p / KW;
                int iy = oy * stride + ky - pad;
                int ix = ox * stride + kx - pad;
                if (iy >= 0 && iy < H && ix >= 0 && ix < W)
                    av = in[(((long)b * H + iy) * W + ix) * Cin + ci];
            }
            a[e] = (__bf16)av;
            int kb = k0 + ((e < 8) ? e : e + 8) + hi * 8;
            float bv = 0.0f;
            if (kb < Ktot && n < Cout)
                bv = w[(long)kb * Cout + n];
            bm[e] = (__bf16)bv;
        }
        acc = __builtin_amdgcn_wmma_f32_16x16x32_bf16(false, a, false, bm,
                                                      (short)0, acc, false, false);
    }

#pragma unroll
    for (int r = 0; r < 8; ++r) {
        int mo = mBase + r + hi * 8;
        int no = nBase + lr;
        if (mo < M && no < Cout)
            out[(long)mo * Cout + no] = acc[r] + bias[no];
    }
}

// ---------------------------------------------------------------------------
// GroupNorm (32 groups) stats: one block per (batch, group)
// ---------------------------------------------------------------------------
__global__ void gn_stats(const float* __restrict__ in, float* __restrict__ stats,
                         int HW, int C)
{
    const int bg = blockIdx.x;
    const int g  = bg & 31;
    const int b  = bg >> 5;
    const int cpg = C >> 5;
    const long n = (long)HW * cpg;
    float s = 0.0f, ss = 0.0f;
    for (long i = threadIdx.x; i < n; i += blockDim.x) {
        long hw = i / cpg;
        int  cc = (int)(i % cpg);
        float v = in[((long)b * HW + hw) * C + g * cpg + cc];
        s += v; ss += v * v;
    }
    __shared__ float rs[256], rss[256];
    rs[threadIdx.x] = s; rss[threadIdx.x] = ss;
    __syncthreads();
    for (int o = blockDim.x >> 1; o > 0; o >>= 1) {
        if ((int)threadIdx.x < o) {
            rs[threadIdx.x]  += rs[threadIdx.x + o];
            rss[threadIdx.x] += rss[threadIdx.x + o];
        }
        __syncthreads();
    }
    if (threadIdx.x == 0) {
        float mu  = rs[0] / (float)n;
        float var = rss[0] / (float)n - mu * mu;
        stats[bg * 2]     = mu;
        stats[bg * 2 + 1] = var;
    }
}

__global__ void gn_silu_k(const float* __restrict__ in, const float* __restrict__ scale,
                          const float* __restrict__ bias, const float* __restrict__ stats,
                          float* __restrict__ out, long total, int HW, int C)
{
    long i = blockIdx.x * (long)blockDim.x + threadIdx.x;
    if (i >= total) return;
    int  c = (int)(i % C);
    int  b = (int)(i / ((long)HW * C));
    int  g = c / (C >> 5);
    float mu  = stats[(b * 32 + g) * 2];
    float var = stats[(b * 32 + g) * 2 + 1];
    float y = (in[i] - mu) * rsqrtf(var + 1e-5f) * scale[c] + bias[c];
    out[i] = y / (1.0f + expf(-y));
}

__global__ void resize2x(const float* __restrict__ in, float* __restrict__ out,
                         int B, int Hin, int Win, int C)
{
    long total = (long)B * Hin * 2 * Win * 2 * C;
    long i = blockIdx.x * (long)blockDim.x + threadIdx.x;
    if (i >= total) return;
    int c = (int)(i % C); long t = i / C;
    int x = (int)(t % (Win * 2)); t /= (Win * 2);
    int y = (int)(t % (Hin * 2));
    int b = (int)(t / (Hin * 2));
    out[i] = in[(((long)b * Hin + (y >> 1)) * Win + (x >> 1)) * C + c];
}

// ---------------------------------------------------------------------------
// VQ
// ---------------------------------------------------------------------------
__global__ void emb_norm_k(const float* __restrict__ emb, float* __restrict__ enorm)
{
    int k = threadIdx.x;                 // 512 threads
    float s = 0.0f;
    for (int d = 0; d < 64; ++d) {
        float v = emb[(long)d * 512 + k];
        s += v * v;
    }
    enorm[k] = s;
}

// scores = |e_n|^2 - 2 z.e ; embT is bf16 [512][64]
__global__ void vq_argmin(const float* __restrict__ z, const __bf16* __restrict__ embT,
                          const float* __restrict__ enorm, int* __restrict__ idx, int M)
{
    const int wave = blockIdx.x * (blockDim.x >> 5) + (threadIdx.x >> 5);
    const int lane = threadIdx.x & 31;
    const int mtiles = M >> 4;
    if (wave >= mtiles) return;
    const int mBase = wave << 4;
    const int hi = lane >> 4;
    const int lr = lane & 15;

    const float* zr = z + (long)(mBase + lr) * 64;
    v16bf a0, a1;
#pragma unroll
    for (int c = 0; c < 2; ++c) {
        float4 p0 = *(const float4*)(zr + c * 32 + hi * 16);
        float4 p1 = *(const float4*)(zr + c * 32 + hi * 16 + 4);
        float4 p2 = *(const float4*)(zr + c * 32 + hi * 16 + 8);
        float4 p3 = *(const float4*)(zr + c * 32 + hi * 16 + 12);
        v16bf a;
        a[0]=(__bf16)p0.x;  a[1]=(__bf16)p0.y;  a[2]=(__bf16)p0.z;  a[3]=(__bf16)p0.w;
        a[4]=(__bf16)p1.x;  a[5]=(__bf16)p1.y;  a[6]=(__bf16)p1.z;  a[7]=(__bf16)p1.w;
        a[8]=(__bf16)p2.x;  a[9]=(__bf16)p2.y;  a[10]=(__bf16)p2.z; a[11]=(__bf16)p2.w;
        a[12]=(__bf16)p3.x; a[13]=(__bf16)p3.y; a[14]=(__bf16)p3.z; a[15]=(__bf16)p3.w;
        if (c == 0) a0 = a; else a1 = a;
    }

    float bestV[8]; int bestI[8];
#pragma unroll
    for (int r = 0; r < 8; ++r) { bestV[r] = 3.4e38f; bestI[r] = 0; }

    for (int nt = 0; nt < 32; ++nt) {
        int n = nt * 16 + lr;
        const __bf16* brow = embT + (long)n * 64;
        v16bf b0 = *(const v16bf*)(brow + hi * 16);
        v16bf b1 = *(const v16bf*)(brow + 32 + hi * 16);
        v8f acc = {};
        acc = __builtin_amdgcn_wmma_f32_16x16x32_bf16(false, a0, false, b0, (short)0, acc, false, false);
        acc = __builtin_amdgcn_wmma_f32_16x16x32_bf16(false, a1, false, b1, (short)0, acc, false, false);
        float en = enorm[n];
#pragma unroll
        for (int r = 0; r < 8; ++r) {
            float v = en - 2.0f * acc[r];
            if (v < bestV[r] || (v == bestV[r] && n < bestI[r])) { bestV[r] = v; bestI[r] = n; }
        }
    }

#pragma unroll
    for (int r = 0; r < 8; ++r) {
        for (int off = 1; off < 16; off <<= 1) {
            float ov = __shfl_xor(bestV[r], off, 32);
            int   oi = __shfl_xor(bestI[r], off, 32);
            if (ov < bestV[r] || (ov == bestV[r] && oi < bestI[r])) { bestV[r] = ov; bestI[r] = oi; }
        }
        if (lr == 0) idx[mBase + r + hi * 8] = bestI[r];
    }
}

__global__ void vq_gather(const float* __restrict__ z, const float* __restrict__ emb,
                          const int* __restrict__ idx, float* __restrict__ q,
                          float* __restrict__ sqacc, long total)
{
    __shared__ float red[256];
    long i = blockIdx.x * (long)blockDim.x + threadIdx.x;
    float d2 = 0.0f;
    if (i < total) {
        int  d = (int)(i & 63);
        long m = i >> 6;
        float qv = emb[(long)d * 512 + idx[m]];
        q[i] = qv;
        float df = qv - z[i];
        d2 = df * df;
    }
    red[threadIdx.x] = d2;
    __syncthreads();
    for (int o = 128; o > 0; o >>= 1) {
        if ((int)threadIdx.x < o) red[threadIdx.x] += red[threadIdx.x + o];
        __syncthreads();
    }
    if (threadIdx.x == 0) atomicAdd(sqacc, red[0]);
}

__global__ void vq_count(const int* __restrict__ idx, float* __restrict__ counts, int M)
{
    int i = blockIdx.x * blockDim.x + threadIdx.x;
    if (i < M) atomicAdd(&counts[idx[i]], 1.0f);
}

__global__ void vq_finalize(const float* __restrict__ sqacc, const float* __restrict__ counts,
                            float* __restrict__ outScalars, int M)
{
    __shared__ float red[512];
    int k = threadIdx.x;
    float avg = counts[k] / (float)M;
    red[k] = avg * logf(avg + 1e-10f);
    __syncthreads();
    for (int o = 256; o > 0; o >>= 1) {
        if (k < o) red[k] += red[k + o];
        __syncthreads();
    }
    if (k == 0) {
        outScalars[0] = 1.25f * sqacc[0] / ((float)M * 64.0f);
        outScalars[1] = expf(-red[0]);
    }
}

// ---------------------------------------------------------------------------
// Host orchestration
// ---------------------------------------------------------------------------
extern "C" void kernel_launch(void* const* d_in, const int* in_sizes, int n_in,
                              void* d_out, int out_size, void* d_ws, size_t ws_size,
                              hipStream_t stream)
{
    if (n_in < 82) return;
    const float* P[82];
    for (int i = 0; i < 82; ++i) P[i] = (const float*)d_in[i];

    // Flat pytree order (dicts sorted by key): params{dec,emb,enc}, x
    const float *dec_in_b = P[0],  *dec_in_w = P[1];
    const float *dec_out_b = P[2], *dec_out_w = P[3];
    const float *drb0_c1b=P[4], *drb0_c1w=P[5], *drb0_c2b=P[6], *drb0_c2w=P[7],
                *drb0_g1b=P[8], *drb0_g1s=P[9], *drb0_g2b=P[10], *drb0_g2s=P[11];
    const float *drb1_c1b=P[12], *drb1_c1w=P[13], *drb1_c2b=P[14], *drb1_c2w=P[15],
                *drb1_g1b=P[16], *drb1_g1s=P[17], *drb1_g2b=P[18], *drb1_g2s=P[19];
    const float *drb2_c1b=P[20], *drb2_c1w=P[21], *drb2_c2b=P[22], *drb2_c2w=P[23],
                *drb2_g1b=P[24], *drb2_g1s=P[25], *drb2_g2b=P[26], *drb2_g2s=P[27],
                *drb2_skb=P[28], *drb2_skw=P[29];
    const float *drb3_c1b=P[30], *drb3_c1w=P[31], *drb3_c2b=P[32], *drb3_c2w=P[33],
                *drb3_g1b=P[34], *drb3_g1s=P[35], *drb3_g2b=P[36], *drb3_g2s=P[37];
    const float *dec_up_b=P[38], *dec_up_w=P[39];
    const float *emb = P[40];
    const float *enc_down_b=P[41], *enc_down_w=P[42];
    const float *enc_in_b=P[43],  *enc_in_w=P[44];
    const float *enc_out_b=P[45], *enc_out_w=P[46];
    const float *erb0_c1b=P[47], *erb0_c1w=P[48], *erb0_c2b=P[49], *erb0_c2w=P[50],
                *erb0_g1b=P[51], *erb0_g1s=P[52], *erb0_g2b=P[53], *erb0_g2s=P[54];
    const float *erb1_c1b=P[55], *erb1_c1w=P[56], *erb1_c2b=P[57], *erb1_c2w=P[58],
                *erb1_g1b=P[59], *erb1_g1s=P[60], *erb1_g2b=P[61], *erb1_g2s=P[62];
    const float *erb2_c1b=P[63], *erb2_c1w=P[64], *erb2_c2b=P[65], *erb2_c2w=P[66],
                *erb2_g1b=P[67], *erb2_g1s=P[68], *erb2_g2b=P[69], *erb2_g2s=P[70],
                *erb2_skb=P[71], *erb2_skw=P[72];
    const float *erb3_c1b=P[73], *erb3_c1w=P[74], *erb3_c2b=P[75], *erb3_c2w=P[76],
                *erb3_g1b=P[77], *erb3_g1s=P[78], *erb3_g2b=P[79], *erb3_g2s=P[80];
    const float *x_in = P[81];

    // ---- workspace layout ----
    char* base = (char*)d_ws;
    const size_t SLOT_BYTES = 4ull * 128 * 128 * 256 * sizeof(float);  // 64 MiB
    float* S0 = (float*)(base);
    float* S1 = (float*)(base + SLOT_BYTES);
    float* S2 = (float*)(base + 2 * SLOT_BYTES);
    float* S3 = (float*)(base + 3 * SLOT_BYTES);
    char*  aux = base + 4 * SLOT_BYTES;
    float* stats  = (float*)(aux);                 // 256 f32
    float* enorm  = (float*)(aux + 4096);          // 512 f32
    float* counts = (float*)(aux + 8192);          // 512 f32
    float* sqacc  = (float*)(aux + 12288);         // 1 f32
    int*   idxbuf = (int*)  (aux + 16384);         // 16384 i32
    __bf16* WT    = (__bf16*)(aux + 16384 + 65536);            // up to 589824 bf16
    __bf16* embT  = (__bf16*)(aux + 16384 + 65536 + 1245184);  // 32768 bf16

    const int Bn = 4;

    auto conv = [&](const float* in, int H, int W, int Cin,
                    const float* wgt, const float* bia, int Cout,
                    int K, int stride, int pad, const float* res,
                    float* out, int Ho, int Wo) {
        int M = Bn * Ho * Wo;
        if ((Cin & 31) == 0) {
            long kc = (long)K * K * Cin * Cout;
            transform_w<<<(int)((kc + 255) / 256), 256, 0, stream>>>(wgt, WT,
                                                                     K * K * Cin, Cout);
            int tiles = ((M + 15) / 16) * ((Cout + 63) / 64);
            int blocks = (tiles + 3) / 4;
            conv_wmma_al<<<blocks, 128, 0, stream>>>(in, WT, bia, res, out,
                                                     Bn, H, W, Cin, Ho, Wo, Cout,
                                                     K, K, stride, pad);
        } else {
            int tiles = ((M + 15) / 16) * ((Cout + 15) / 16);
            int blocks = (tiles + 3) / 4;
            conv_wmma_gen<<<blocks, 128, 0, stream>>>(in, wgt, bia, out,
                                                      Bn, H, W, Cin, Ho, Wo, Cout,
                                                      K, K, stride, pad);
        }
    };

    auto gnsilu = [&](const float* in, const float* s, const float* bb,
                      float* out, int HW, int C) {
        gn_stats<<<Bn * 32, 256, 0, stream>>>(in, stats, HW, C);
        long total = (long)Bn * HW * C;
        gn_silu_k<<<(int)((total + 255) / 256), 256, 0, stream>>>(in, s, bb, stats, out,
                                                                  total, HW, C);
    };

    auto resblock = [&](float* xb, float* tA, float* tB, float* tC,
                        int H, int W, int Cin, int Cout,
                        const float* g1b, const float* g1s,
                        const float* c1b, const float* c1w,
                        const float* g2b, const float* g2s,
                        const float* c2b, const float* c2w,
                        const float* skb, const float* skw) -> float* {
        int HW = H * W;
        gnsilu(xb, g1s, g1b, tA, HW, Cin);
        conv(tA, H, W, Cin, c1w, c1b, Cout, 3, 1, 1, nullptr, tB, H, W);
        gnsilu(tB, g2s, g2b, tA, HW, Cout);
        const float* res;
        float* out;
        if (skw) {
            conv(xb, H, W, Cin, skw, skb, Cout, 1, 1, 0, nullptr, tC, H, W);
            res = tC; out = xb;
        } else {
            res = xb; out = tB;
        }
        conv(tA, H, W, Cout, c2w, c2b, Cout, 3, 1, 1, res, out, H, W);
        return out;
    };

    // ---------------- encoder ----------------
    conv(x_in, 128, 128, 3, enc_in_w, enc_in_b, 128, 3, 1, 1, nullptr, S0, 128, 128);
    float* h = resblock(S0, S1, S2, S3, 128, 128, 128, 128,
                        erb0_g1b, erb0_g1s, erb0_c1b, erb0_c1w,
                        erb0_g2b, erb0_g2s, erb0_c2b, erb0_c2w, nullptr, nullptr);   // -> S2
    h = resblock(h, S1, S0, S3, 128, 128, 128, 128,
                 erb1_g1b, erb1_g1s, erb1_c1b, erb1_c1w,
                 erb1_g2b, erb1_g2s, erb1_c2b, erb1_c2w, nullptr, nullptr);          // -> S0
    conv(h, 128, 128, 128, enc_down_w, enc_down_b, 128, 3, 2, 1, nullptr, S1, 64, 64);
    h = resblock(S1, S0, S2, S3, 64, 64, 128, 256,
                 erb2_g1b, erb2_g1s, erb2_c1b, erb2_c1w,
                 erb2_g2b, erb2_g2s, erb2_c2b, erb2_c2w, erb2_skb, erb2_skw);        // -> S1
    h = resblock(h, S0, S2, S3, 64, 64, 256, 256,
                 erb3_g1b, erb3_g1s, erb3_c1b, erb3_c1w,
                 erb3_g2b, erb3_g2s, erb3_c2b, erb3_c2w, nullptr, nullptr);          // -> S2
    conv(h, 64, 64, 256, enc_out_w, enc_out_b, 64, 3, 1, 1, nullptr, S0, 64, 64);    // z -> S0

    // ---------------- VQ ----------------
    const int M = Bn * 64 * 64;               // 16384 latent pixels
    emb_norm_k<<<1, 512, 0, stream>>>(emb, enorm);
    transform_w<<<(64 * 512 + 255) / 256, 256, 0, stream>>>(emb, embT, 64, 512);
    hipMemsetAsync(counts, 0, 512 * sizeof(float), stream);
    hipMemsetAsync(sqacc, 0, sizeof(float), stream);
    vq_argmin<<<(M / 16 + 3) / 4, 128, 0, stream>>>(S0, embT, enorm, idxbuf, M);
    vq_gather<<<(int)(((long)M * 64 + 255) / 256), 256, 0, stream>>>(S0, emb, idxbuf, S1,
                                                                     sqacc, (long)M * 64);
    vq_count<<<(M + 255) / 256, 256, 0, stream>>>(idxbuf, counts, M);
    vq_finalize<<<1, 512, 0, stream>>>(sqacc, counts, (float*)d_out + 196608, M);

    // ---------------- decoder ----------------
    conv(S1, 64, 64, 64, dec_in_w, dec_in_b, 256, 3, 1, 1, nullptr, S2, 64, 64);
    h = resblock(S2, S0, S1, S3, 64, 64, 256, 256,
                 drb0_g1b, drb0_g1s, drb0_c1b, drb0_c1w,
                 drb0_g2b, drb0_g2s, drb0_c2b, drb0_c2w, nullptr, nullptr);          // -> S1
    h = resblock(h, S0, S2, S3, 64, 64, 256, 256,
                 drb1_g1b, drb1_g1s, drb1_c1b, drb1_c1w,
                 drb1_g2b, drb1_g2s, drb1_c2b, drb1_c2w, nullptr, nullptr);          // -> S2
    {
        long total = (long)Bn * 128 * 128 * 256;
        resize2x<<<(int)((total + 255) / 256), 256, 0, stream>>>(h, S0, Bn, 64, 64, 256);
    }
    conv(S0, 128, 128, 256, dec_up_w, dec_up_b, 256, 3, 1, 1, nullptr, S1, 128, 128);
    h = resblock(S1, S0, S2, S3, 128, 128, 256, 128,
                 drb2_g1b, drb2_g1s, drb2_c1b, drb2_c1w,
                 drb2_g2b, drb2_g2s, drb2_c2b, drb2_c2w, drb2_skb, drb2_skw);        // -> S1
    h = resblock(h, S0, S2, S3, 128, 128, 128, 128,
                 drb3_g1b, drb3_g1s, drb3_c1b, drb3_c1w,
                 drb3_g2b, drb3_g2s, drb3_c2b, drb3_c2w, nullptr, nullptr);          // -> S2
    conv(h, 128, 128, 128, dec_out_w, dec_out_b, 3, 3, 1, 1, nullptr,
         (float*)d_out, 128, 128);            // recon -> d_out[0..196607]
    (void)in_sizes; (void)out_size; (void)ws_size;
}
